// MultiHeadCoAttention_28174985462288
// MI455X (gfx1250) — matrix operand order repaired and checked by reading, not compiled, not run on hardware
//
#include <hip/hip_runtime.h>
#include <hip/hip_bf16.h>
#include <math.h>

// ---------------- problem constants ----------------
#define BATCH 1024
#define SEQ   64
#define DIM   512
#define NHEAD 8
#define DHEAD 64
#define SCALE 0.044194173824159216f   // 1/sqrt(512)

// LDS strides (elements) chosen to avoid bank conflicts
#define ATT_LDW 65   // fp32 attn matrix
#define BF_LDW  72   // bf16 tiles (stride 144B, 16B-aligned)

typedef __attribute__((ext_vector_type(16))) __bf16 v16bf;
typedef __attribute__((ext_vector_type(8)))  __bf16 v8bf;
typedef __attribute__((ext_vector_type(8)))  float  v8f;

// ---------------- helpers ----------------
__device__ __forceinline__ v8f v8f_zero() {
    v8f z;
#pragma unroll
    for (int i = 0; i < 8; ++i) z[i] = 0.0f;
    return z;
}

__device__ __forceinline__ v8f wmma_bf16(v16bf a, v16bf b, v8f c) {
    // D(16x16,f32) = A(16x32,bf16) * B(32x16,bf16) + C
    return __builtin_amdgcn_wmma_f32_16x16x32_bf16(
        /*neg_a=*/false, a, /*neg_b=*/false, b,
        /*c_mod=*/(short)0, c, /*reuse_a=*/false, /*reuse_b=*/false);
}

// A fragment (16x32 bf16) from row-major bf16 source.
// Lane l (half=l>>4, m=l&15): row row0+m, K chunks [k0+8h .. +7] and [k0+16+8h .. +7]
__device__ __forceinline__ v16bf load_a_bf16(const __bf16* base, int ldw,
                                             int row0, int k0, int lane) {
    const int half = lane >> 4, m = lane & 15;
    const __bf16* p = base + (row0 + m) * ldw + k0 + half * 8;
    v8bf c0 = *(const v8bf*)(p);
    v8bf c1 = *(const v8bf*)(p + 16);
    v16bf a;
#pragma unroll
    for (int i = 0; i < 8; ++i) { a[i] = c0[i]; a[i + 8] = c1[i]; }
    return a;
}

// A fragment built from fp32 row-major source (convert to bf16 in regs)
__device__ __forceinline__ v16bf load_a_f32(const float* base, int ldw,
                                            int row0, int k0, int lane) {
    const int half = lane >> 4, m = lane & 15;
    const float* p = base + (row0 + m) * ldw + k0 + half * 8;
    v8f c0 = *(const v8f*)(p);
    v8f c1 = *(const v8f*)(p + 16);
    v16bf a;
#pragma unroll
    for (int i = 0; i < 8; ++i) { a[i] = (__bf16)c0[i]; a[i + 8] = (__bf16)c1[i]; }
    return a;
}

// B fragment (32x16 bf16) from K-contiguous ("transposed", [n][k]) bf16 source.
// Lane l (half=l>>4, n=l&15): column n0+n, K chunks [k0+8h .. +7] and [k0+16+8h .. +7]
__device__ __forceinline__ v16bf load_b_bf16(const __bf16* base, int ldw,
                                             int k0, int n0, int lane) {
    const int half = lane >> 4, n = lane & 15;
    const __bf16* p = base + (n0 + n) * ldw + k0 + half * 8;
    v8bf c0 = *(const v8bf*)(p);
    v8bf c1 = *(const v8bf*)(p + 16);
    v16bf b;
#pragma unroll
    for (int i = 0; i < 8; ++i) { b[i] = c0[i]; b[i + 8] = c1[i]; }
    return b;
}

// ---------------- prep: transpose+convert weights to bf16 [n][k] ----------------
__global__ void prep_weights_kernel(const float* __restrict__ Wq,
                                    const float* __restrict__ Wc,
                                    __bf16* __restrict__ WqT,
                                    __bf16* __restrict__ WcT) {
    int idx = blockIdx.x * blockDim.x + threadIdx.x;   // 0 .. 512*512-1
    int n = idx >> 9;
    int k = idx & 511;
    WqT[idx] = (__bf16)Wq[k * DIM + n];
    WcT[idx] = (__bf16)Wc[k * DIM + n];
}

// ---------------- fused co-attention, one block per (b,h) ----------------
__global__ __launch_bounds__(128, 1)
void coattn_kernel(const float* __restrict__ query,
                   const float* __restrict__ context,
                   const __bf16* __restrict__ WqT,
                   const __bf16* __restrict__ WcT,
                   const float* __restrict__ bq,
                   const float* __restrict__ bc,
                   float* __restrict__ c_out,
                   float* __restrict__ q_out) {
    __shared__ float  attn  [SEQ * ATT_LDW];   // 16.6 KB
    __shared__ __bf16 qh    [SEQ * BF_LDW];    // q_h  [q][d]
    __shared__ __bf16 qhT   [SEQ * BF_LDW];    // q_h^T[d][q]
    __shared__ __bf16 ch    [SEQ * BF_LDW];    // c_h  [j][d]
    __shared__ __bf16 cattn [SEQ * BF_LDW];    // softmax_k(attn)   [q][k]
    __shared__ __bf16 qattnT[SEQ * BF_LDW];    // softmax_q(attn)^T [k][q]

    const int h    = blockIdx.x;
    const int b    = blockIdx.y;
    const int tid  = threadIdx.x;
    const int wave = tid >> 5;
    const int lane = tid & 31;
    const int half = lane >> 4;
    const int lm   = lane & 15;
    const int row0 = wave * 16;              // this wave's 16-row strip

    const float*  qbase = query   + (size_t)b * SEQ * DIM;
    const float*  cbase = context + (size_t)b * SEQ * DIM;
    const __bf16* wqh   = WqT + (size_t)h * DHEAD * DIM;   // [n][k], n local 0..63
    const __bf16* wch   = WcT + (size_t)h * DHEAD * DIM;

    // ---- Phase 1: q_h strip = query_b[row0:+16, :] @ Wq_h  (K=512) ----
    v8f acc[4];
#pragma unroll
    for (int nt = 0; nt < 4; ++nt) acc[nt] = v8f_zero();
    for (int k0 = 0; k0 < DIM; k0 += 32) {
        v16bf a = load_a_f32(qbase, DIM, row0, k0, lane);
#pragma unroll
        for (int nt = 0; nt < 4; ++nt) {
            v16bf bf = load_b_bf16(wqh, DIM, k0, nt * 16, lane);
            acc[nt] = wmma_bf16(a, bf, acc[nt]);
        }
    }
#pragma unroll
    for (int nt = 0; nt < 4; ++nt) {
        int col = nt * 16 + lm;
        float bias = bq[h * DHEAD + col];
#pragma unroll
        for (int r = 0; r < 8; ++r) {
            int row = row0 + r + half * 8;
            __bf16 v = (__bf16)(acc[nt][r] + bias);
            qh [row * BF_LDW + col] = v;
            qhT[col * BF_LDW + row] = v;
        }
    }

    // ---- Phase 2: c_h strip = context_b[row0:+16, :] @ Wc_h ----
#pragma unroll
    for (int nt = 0; nt < 4; ++nt) acc[nt] = v8f_zero();
    for (int k0 = 0; k0 < DIM; k0 += 32) {
        v16bf a = load_a_f32(cbase, DIM, row0, k0, lane);
#pragma unroll
        for (int nt = 0; nt < 4; ++nt) {
            v16bf bf = load_b_bf16(wch, DIM, k0, nt * 16, lane);
            acc[nt] = wmma_bf16(a, bf, acc[nt]);
        }
    }
#pragma unroll
    for (int nt = 0; nt < 4; ++nt) {
        int col = nt * 16 + lm;
        float bias = bc[h * DHEAD + col];
#pragma unroll
        for (int r = 0; r < 8; ++r)
            ch[(row0 + r + half * 8) * BF_LDW + col] = (__bf16)(acc[nt][r] + bias);
    }
    __syncthreads();

    // ---- Phase 3: attn strip = q_h @ c_h^T * SCALE ----
#pragma unroll
    for (int nt = 0; nt < 4; ++nt) acc[nt] = v8f_zero();
#pragma unroll
    for (int k0 = 0; k0 < DHEAD; k0 += 32) {
        v16bf a = load_a_bf16(qh, BF_LDW, row0, k0, lane);
#pragma unroll
        for (int nt = 0; nt < 4; ++nt) {
            v16bf bf = load_b_bf16(ch, BF_LDW, k0, nt * 16, lane);  // c_h^T: col j = row j of c_h
            acc[nt] = wmma_bf16(a, bf, acc[nt]);
        }
    }
#pragma unroll
    for (int nt = 0; nt < 4; ++nt) {
        int col = nt * 16 + lm;
#pragma unroll
        for (int r = 0; r < 8; ++r)
            attn[(row0 + r + half * 8) * ATT_LDW + col] = acc[nt][r] * SCALE;
    }
    __syncthreads();

    // ---- Phase 4: dual softmax (waves 0-1: rows -> c_attn; waves 2-3: cols -> q_attn^T) ----
    if (tid < 64) {
        const int r = tid;
        float m = -INFINITY;
        for (int k = 0; k < SEQ; ++k) m = fmaxf(m, attn[r * ATT_LDW + k]);
        float s = 0.0f;
        for (int k = 0; k < SEQ; ++k) s += __expf(attn[r * ATT_LDW + k] - m);
        float inv = 1.0f / s;
        for (int k = 0; k < SEQ; ++k)
            cattn[r * BF_LDW + k] = (__bf16)(__expf(attn[r * ATT_LDW + k] - m) * inv);
    } else {
        const int k = tid - 64;
        float m = -INFINITY;
        for (int q = 0; q < SEQ; ++q) m = fmaxf(m, attn[q * ATT_LDW + k]);
        float s = 0.0f;
        for (int q = 0; q < SEQ; ++q) s += __expf(attn[q * ATT_LDW + k] - m);
        float inv = 1.0f / s;
        for (int q = 0; q < SEQ; ++q)
            qattnT[k * BF_LDW + q] = (__bf16)(__expf(attn[q * ATT_LDW + k] - m) * inv);
    }
    __syncthreads();

    // ---- Phase 5: c_co strip = c_attn @ c_h^T  -> c_out[b, :, h*64:] ----
#pragma unroll
    for (int nt = 0; nt < 4; ++nt) acc[nt] = v8f_zero();
#pragma unroll
    for (int k0 = 0; k0 < SEQ; k0 += 32) {
        v16bf a = load_a_bf16(cattn, BF_LDW, row0, k0, lane);
#pragma unroll
        for (int nt = 0; nt < 4; ++nt) {
            v16bf bf = load_b_bf16(ch, BF_LDW, k0, nt * 16, lane);
            acc[nt] = wmma_bf16(a, bf, acc[nt]);
        }
    }
    {
        float* cob = c_out + (size_t)b * SEQ * DIM + (size_t)h * DHEAD;
#pragma unroll
        for (int nt = 0; nt < 4; ++nt) {
            int col = nt * 16 + lm;
#pragma unroll
            for (int r = 0; r < 8; ++r)
                cob[(size_t)(row0 + r + half * 8) * DIM + col] = acc[nt][r];
        }
    }

    // ---- Phase 6: q_co strip = q_attn^T @ q_h  -> q_out[b, :, h*64:] ----
#pragma unroll
    for (int nt = 0; nt < 4; ++nt) acc[nt] = v8f_zero();
#pragma unroll
    for (int k0 = 0; k0 < SEQ; k0 += 32) {
        v16bf a = load_a_bf16(qattnT, BF_LDW, row0, k0, lane);   // A[m=k][q]
#pragma unroll
        for (int nt = 0; nt < 4; ++nt) {
            v16bf bf = load_b_bf16(qhT, BF_LDW, k0, nt * 16, lane); // B[k=q][d] via q_h^T
            acc[nt] = wmma_bf16(a, bf, acc[nt]);
        }
    }
    {
        float* qob = q_out + (size_t)b * SEQ * DIM + (size_t)h * DHEAD;
#pragma unroll
        for (int nt = 0; nt < 4; ++nt) {
            int col = nt * 16 + lm;
#pragma unroll
            for (int r = 0; r < 8; ++r)
                qob[(size_t)(row0 + r + half * 8) * DIM + col] = acc[nt][r];
        }
    }
}

// ---------------- launch ----------------
extern "C" void kernel_launch(void* const* d_in, const int* in_sizes, int n_in,
                              void* d_out, int out_size, void* d_ws, size_t ws_size,
                              hipStream_t stream) {
    const float* query   = (const float*)d_in[0];
    const float* context = (const float*)d_in[1];
    const float* Wq      = (const float*)d_in[2];
    const float* bq      = (const float*)d_in[3];
    const float* Wc      = (const float*)d_in[4];
    const float* bc      = (const float*)d_in[5];

    float* c_out = (float*)d_out;
    float* q_out = c_out + (size_t)BATCH * SEQ * DIM;

    __bf16* WqT = (__bf16*)d_ws;                    // 512*512 bf16 = 512 KB
    __bf16* WcT = WqT + (size_t)DIM * DIM;          // +512 KB (ws total 1 MB)

    prep_weights_kernel<<<(DIM * DIM) / 256, 256, 0, stream>>>(Wq, Wc, WqT, WcT);

    dim3 grid(NHEAD, BATCH);   // 8192 blocks, one per (b,h)
    coattn_kernel<<<grid, 128, 0, stream>>>(query, context, WqT, WcT, bq, bc, c_out, q_out);
}
